// MetaNodeRNN_59193239273820
// MI455X (gfx1250) — compile-verified, matrix operand order checked by reading, not run on hardware
//
#include <hip/hip_runtime.h>
#include <hip/hip_bf16.h>

// Problem dims (from reference)
#define B_  256
#define E_  256
#define H_  256
#define M_  128
#define T_  512

// Persistent-kernel shape: 48 WGs x 8 waves = 384 waves -> exactly one
// 16x16 tile per wave per GRU step (256 gate tiles + 128 proj tiles).
#define G_WG 48
#define TPB  256
#define WPG  8
#define GW   (G_WG * WPG)

typedef __bf16 bf16;
typedef bf16  v16bf __attribute__((ext_vector_type(16)));
typedef float v8f   __attribute__((ext_vector_type(8)));
typedef unsigned short u16;

// ---------------------------------------------------------------------------
// 16-bit A/B fragment load per CDNA5 ISA 7.12.2:
// lane L holds row (L%16); L/16 selects K sub-chunks {8h..8h+7} and {16+8h..}.
// Two 16-byte contiguous loads per fragment (row-major, k contiguous).
// ---------------------------------------------------------------------------
struct alignas(16) Chunk2 { uint4 lo, hi; };

__device__ __forceinline__ v16bf load_frag(const u16* __restrict__ rowbase,
                                           int k0, int half) {
  Chunk2 c;
  c.lo = *(const uint4*)(rowbase + k0 + 8 * half);
  c.hi = *(const uint4*)(rowbase + k0 + 16 + 8 * half);
  return __builtin_bit_cast(v16bf, c);
}

__device__ __forceinline__ v8f wmma_bf(v16bf a, v16bf b, v8f c) {
  // (neg_a, A, neg_b, B, c_mod, C, reuse_a, reuse_b)
  return __builtin_amdgcn_wmma_f32_16x16x32_bf16(false, a, false, b,
                                                 (short)0, c, false, false);
}

__device__ __forceinline__ float sigmf_(float x) {
  return 1.0f / (1.0f + __expf(-x));
}
__device__ __forceinline__ float tanhf_(float x) {
  x = fminf(fmaxf(x, -15.0f), 15.0f);
  float e = __expf(2.0f * x);
  return (e - 1.0f) / (e + 1.0f);
}
__device__ __forceinline__ u16 f2bf(float x) {
  return __builtin_bit_cast(u16, (bf16)x);
}

// Device-wide step barrier: monotone counter in L2, release/acquire fences.
__device__ __forceinline__ void grid_barrier(int* cnt, int barn) {
  __threadfence();              // release our stores to device scope
  __syncthreads();
  if (threadIdx.x == 0) {
    atomicAdd(cnt, 1);
    while (__hip_atomic_load(cnt, __ATOMIC_RELAXED,
                             __HIP_MEMORY_SCOPE_AGENT) < G_WG * barn) {
      __builtin_amdgcn_s_sleep(1);
    }
  }
  __syncthreads();
  __threadfence();              // acquire other WGPs' stores
}

// Plain GEMM tile with two independent half-K accumulator chains (more ILP
// for the in-order WMMA pipe), K = 256.
__device__ __forceinline__ v8f gemm_tile_k256(const u16* __restrict__ arow,
                                              const u16* __restrict__ brow,
                                              int half) {
  v8f a0 = {}, a1 = {};
#pragma unroll
  for (int k0 = 0; k0 < 256; k0 += 64) {
    a0 = wmma_bf(load_frag(arow, k0, half), load_frag(brow, k0, half), a0);
    a1 = wmma_bf(load_frag(arow, k0 + 32, half), load_frag(brow, k0 + 32, half),
                 a1);
  }
  return a0 + a1;
}

// ---------------------------------------------------------------------------
// Prologue kernels
// ---------------------------------------------------------------------------
__global__ void cvt_bf16_kernel(const float* __restrict__ src,
                                u16* __restrict__ dst, int n) {
  int i = blockIdx.x * blockDim.x + threadIdx.x;
  if (i < n) dst[i] = f2bf(src[i]);
}

__global__ void init_bar_kernel(int* bar) { *bar = 0; }

// ---------------------------------------------------------------------------
// Persistent fused FNN + GRU(512 steps) + out-projection kernel
// ---------------------------------------------------------------------------
__global__ __launch_bounds__(TPB, 1) void rnn_kernel(
    const u16* __restrict__ embb,
    const u16* __restrict__ w1b,  const float* __restrict__ b1,
    const u16* __restrict__ w2b,  const float* __restrict__ b2,
    const u16* __restrict__ wihb, const float* __restrict__ bih,
    const u16* __restrict__ whhb, const float* __restrict__ bhh,
    const u16* __restrict__ outwb, const float* __restrict__ outb,
    u16* __restrict__ tmp1,
    u16* __restrict__ hbf0, u16* __restrict__ hbf1,
    float* __restrict__ hf0, float* __restrict__ hf1,
    int* __restrict__ bar, float* __restrict__ out) {
  const int lane = threadIdx.x & 31;
  const int ln   = lane & 15;     // row/col within tile this lane owns
  const int half = lane >> 4;     // K sub-chunk selector / M-row offset (+8)
  const int gw   = blockIdx.x * WPG + (threadIdx.x >> 5);
  int barn = 0;

  u16*   hbf[2] = {hbf0, hbf1};
  float* hf [2] = {hf0,  hf1 };

  // ---- FNN layer 1: tmp1 = relu(emb @ w1.T + b1)  (bf16) ----
  for (int tt = gw; tt < 256; tt += GW) {
    const int m0 = (tt >> 4) * 16, c0 = (tt & 15) * 16;
    const v8f acc = gemm_tile_k256(embb + (size_t)(m0 + ln) * E_,
                                   w1b + (size_t)(c0 + ln) * E_, half);
    const float bias = b1[c0 + ln];
#pragma unroll
    for (int r = 0; r < 8; ++r) {
      float v = acc[r] + bias;
      v = v > 0.0f ? v : 0.0f;
      tmp1[(size_t)(m0 + r + 8 * half) * H_ + c0 + ln] = f2bf(v);
    }
  }
  grid_barrier(bar, ++barn);

  // ---- FNN layer 2: h0 = tmp1 @ w2.T + b2  (store f32 + bf16) ----
  for (int tt = gw; tt < 256; tt += GW) {
    const int m0 = (tt >> 4) * 16, c0 = (tt & 15) * 16;
    const v8f acc = gemm_tile_k256(tmp1 + (size_t)(m0 + ln) * H_,
                                   w2b + (size_t)(c0 + ln) * H_, half);
    const float bias = b2[c0 + ln];
#pragma unroll
    for (int r = 0; r < 8; ++r) {
      const float v = acc[r] + bias;
      const size_t idx = (size_t)(m0 + r + 8 * half) * H_ + c0 + ln;
      hf[0][idx]  = v;
      hbf[0][idx] = f2bf(v);
    }
  }
  grid_barrier(bar, ++barn);

  // ---- GRU scan: x_t == h_t for t>=1, so ONE A fragment feeds 6 chains ----
  int cur = 0;
  for (int t = 0; t < T_; ++t) {
    const bool first = (t == 0);
    const u16*   hb  = hbf[cur];
    const float* hfc = hf[cur];
    u16*   hbn = hbf[cur ^ 1];
    float* hfn = hf[cur ^ 1];
    const int ntiles = first ? 256 : 384;   // proj of h_{t-1} pipelined in

    for (int tt = gw; tt < ntiles; tt += GW) {
      if (tt < 256) {
        // -------- gate tile: h_new[m0:m0+16, c0:c0+16] --------
        const int m0 = (tt >> 4) * 16, c0 = (tt & 15) * 16;
        const u16* arow = hb + (size_t)(m0 + ln) * H_;
        const u16* wr = whhb + (size_t)(c0 + ln) * H_;
        const u16* wz = whhb + (size_t)(H_ + c0 + ln) * H_;
        const u16* wn = whhb + (size_t)(2 * H_ + c0 + ln) * H_;
        const u16* vr = wihb + (size_t)(c0 + ln) * H_;
        const u16* vz = wihb + (size_t)(H_ + c0 + ln) * H_;
        const u16* vn = wihb + (size_t)(2 * H_ + c0 + ln) * H_;
        v8f ghr = {}, ghz = {}, ghn = {};
        v8f gir = {}, giz = {}, gin = {};
        if (first) {            // t==0: x==0 -> gi is bias-only
#pragma unroll
          for (int k0 = 0; k0 < H_; k0 += 32) {
            const v16bf a = load_frag(arow, k0, half);
            ghr = wmma_bf(a, load_frag(wr, k0, half), ghr);
            ghz = wmma_bf(a, load_frag(wz, k0, half), ghz);
            ghn = wmma_bf(a, load_frag(wn, k0, half), ghn);
          }
        } else {                // merged loop: shared A, 6 indep chains
#pragma unroll
          for (int k0 = 0; k0 < H_; k0 += 32) {
            const v16bf a = load_frag(arow, k0, half);
            ghr = wmma_bf(a, load_frag(wr, k0, half), ghr);
            ghz = wmma_bf(a, load_frag(wz, k0, half), ghz);
            ghn = wmma_bf(a, load_frag(wn, k0, half), ghn);
            gir = wmma_bf(a, load_frag(vr, k0, half), gir);
            giz = wmma_bf(a, load_frag(vz, k0, half), giz);
            gin = wmma_bf(a, load_frag(vn, k0, half), gin);
          }
        }
        const int col = c0 + ln;
        const float bir = bih[col],          bhr = bhh[col];
        const float biz = bih[H_ + col],     bhz = bhh[H_ + col];
        const float bin = bih[2 * H_ + col], bhn = bhh[2 * H_ + col];
#pragma unroll
        for (int r = 0; r < 8; ++r) {
          const int m = m0 + r + 8 * half;
          const float rr = sigmf_(gir[r] + bir + ghr[r] + bhr);
          const float zz = sigmf_(giz[r] + biz + ghz[r] + bhz);
          const float nn = tanhf_(gin[r] + bin + rr * (ghn[r] + bhn));
          const float hp = hfc[(size_t)m * H_ + col];
          const float hv = (1.0f - zz) * nn + zz * hp;
          hfn[(size_t)m * H_ + col] = hv;
          hbn[(size_t)m * H_ + col] = f2bf(hv);
        }
      } else {
        // -------- projection of h_{t-1} (== current state hb) --------
        const int p  = tt - 256;
        const int b0 = (p >> 3) * 16, mo = (p & 7) * 16;
        const v8f acc = gemm_tile_k256(hb + (size_t)(b0 + ln) * H_,
                                       outwb + (size_t)(mo + ln) * H_, half);
        const float bias = outb[mo + ln];
#pragma unroll
        for (int r = 0; r < 8; ++r) {
          const int bb = b0 + r + 8 * half;
          out[(size_t)bb * ((size_t)T_ * M_) + (size_t)(t - 1) * M_ + mo + ln] =
              acc[r] + bias;
        }
      }
    }
    grid_barrier(bar, ++barn);
    cur ^= 1;
  }

  // ---- epilogue: project h_{T-1}; emit hx_final ----
  {
    const u16*   hb  = hbf[cur];
    const float* hfc = hf[cur];
    for (int p = gw; p < 128; p += GW) {
      const int b0 = (p >> 3) * 16, mo = (p & 7) * 16;
      const v8f acc = gemm_tile_k256(hb + (size_t)(b0 + ln) * H_,
                                     outwb + (size_t)(mo + ln) * H_, half);
      const float bias = outb[mo + ln];
#pragma unroll
      for (int r = 0; r < 8; ++r) {
        const int bb = b0 + r + 8 * half;
        out[(size_t)bb * ((size_t)T_ * M_) + (size_t)(T_ - 1) * M_ + mo + ln] =
            acc[r] + bias;
      }
    }
    float* hxout = out + (size_t)B_ * T_ * M_;
    const int gid = blockIdx.x * TPB + threadIdx.x;
    for (int i = gid; i < B_ * H_; i += G_WG * TPB) hxout[i] = hfc[i];
  }
}

// ---------------------------------------------------------------------------
// Host launcher
// ---------------------------------------------------------------------------
extern "C" void kernel_launch(void* const* d_in, const int* in_sizes, int n_in,
                              void* d_out, int out_size, void* d_ws,
                              size_t ws_size, hipStream_t stream) {
  (void)in_sizes; (void)n_in; (void)out_size; (void)ws_size;

  const float* emb  = (const float*)d_in[0];
  const float* w1   = (const float*)d_in[1];
  const float* b1   = (const float*)d_in[2];
  const float* w2   = (const float*)d_in[3];
  const float* b2   = (const float*)d_in[4];
  const float* wih  = (const float*)d_in[5];
  const float* bih  = (const float*)d_in[6];
  const float* whh  = (const float*)d_in[7];
  const float* bhh  = (const float*)d_in[8];
  const float* outw = (const float*)d_in[9];
  const float* outb = (const float*)d_in[10];
  float* out = (float*)d_out;

  // Workspace layout (256B aligned slabs)
  char* ws = (char*)d_ws;
  size_t off = 0;
  auto take = [&](size_t bytes) -> void* {
    void* p = ws + off;
    off += (bytes + 255) & ~(size_t)255;
    return p;
  };
  int*   bar   = (int*)take(256);
  u16*   hbf0  = (u16*)take((size_t)B_ * H_ * 2);
  u16*   hbf1  = (u16*)take((size_t)B_ * H_ * 2);
  float* hf0   = (float*)take((size_t)B_ * H_ * 4);
  float* hf1   = (float*)take((size_t)B_ * H_ * 4);
  u16*   embb  = (u16*)take((size_t)B_ * E_ * 2);
  u16*   w1b   = (u16*)take((size_t)H_ * E_ * 2);
  u16*   w2b   = (u16*)take((size_t)H_ * H_ * 2);
  u16*   wihb  = (u16*)take((size_t)3 * H_ * H_ * 2);
  u16*   whhb  = (u16*)take((size_t)3 * H_ * H_ * 2);
  u16*   outwb = (u16*)take((size_t)M_ * H_ * 2);
  u16*   tmp1  = (u16*)take((size_t)B_ * H_ * 2);

  auto cvt = [&](const float* s, u16* d, int n) {
    cvt_bf16_kernel<<<(n + TPB - 1) / TPB, TPB, 0, stream>>>(s, d, n);
  };
  cvt(emb,  embb,  B_ * E_);
  cvt(w1,   w1b,   H_ * E_);
  cvt(w2,   w2b,   H_ * H_);
  cvt(wih,  wihb,  3 * H_ * H_);
  cvt(whh,  whhb,  3 * H_ * H_);
  cvt(outw, outwb, M_ * H_);
  init_bar_kernel<<<1, 1, 0, stream>>>(bar);

  rnn_kernel<<<G_WG, TPB, 0, stream>>>(embb, w1b, b1, w2b, b2, wihb, bih, whhb,
                                       bhh, outwb, outb, tmp1, hbf0, hbf1, hf0,
                                       hf1, bar, out);
}
